// DenoiseStepOnnxWrapper_41961830482499
// MI455X (gfx1250) — matrix operand order repaired and checked by reading, not compiled, not run on hardware
//
#include <hip/hip_runtime.h>
#include <math.h>

// ---------------------------------------------------------------------------
// DiT denoise step for MI455X (gfx1250, wave32).
// All GEMMs run through V_WMMA_F32_16X16X32_BF16. fp32 operands are staged
// through LDS directly in the WMMA fragment layout (lane-major), so each lane
// fetches its fragment with contiguous ds_load_b128; global traffic uses
// exec-masked float4 (global_load_b128); fp32->bf16 uses the native hardware
// convert via __bf16 casts. 64x64x32 block tiles, 8 waves, double-buffered
// LDS (one barrier per k-step). The KV/score length is padded 850->896 so
// every GEMM has K%32==0 and 16B-aligned rows: no scalar tails.
// ~1.6 GB fp32 traffic/call -> ~70us HBM floor @ 23.3 TB/s.
// ---------------------------------------------------------------------------

#define LL 18
#define HH 1024
#define NH 8
#define HD 256
#define MLPD 4096
#define BB 8
#define SS 50
#define PP 800
#define AD 32
#define TT (BB * SS)      // 400 tokens
#define PT (PP + SS)      // 850 real kv length
#define SCP 896           // padded kv length (28*32, 14*64)

typedef __attribute__((ext_vector_type(16))) __bf16 bf16x16;
typedef __attribute__((ext_vector_type(16))) short  s16x16;
typedef __attribute__((ext_vector_type(8)))  float  f32x8;

// ---- helpers ---------------------------------------------------------------

// Native fp32->bf16 convert (v_cvt_*bf16* on gfx1250).
__device__ __forceinline__ unsigned short f32_to_bf16(float f) {
    __bf16 b = (__bf16)f;
    return __builtin_bit_cast(unsigned short, b);
}

__device__ __forceinline__ unsigned long long pack4_bf16(float4 v) {
    return (unsigned long long)f32_to_bf16(v.x)
         | ((unsigned long long)f32_to_bf16(v.y) << 16)
         | ((unsigned long long)f32_to_bf16(v.z) << 32)
         | ((unsigned long long)f32_to_bf16(v.w) << 48);
}

// LDS tile layout: [subtile 0..3][lane 0..31][e 0..15] unsigned short.
// A (16x32 per subtile, subtile = m/16): lane = m%16 + (k>=16)*16,
//                                        e = k&15
// B (32x16 per subtile, subtile = n/16): lane = n%16 + (k>=16)*16, e = k&15
#define TILE_HALVES (4 * 32 * 16)   // 2048 halves = 4 KB

// ---- tile staging (float4-granular, exec-masked b128 loads) ----------------
// Requires: K % 32 == 0, lda/ldb % 4 == 0, N % 4 == 0, 16B-aligned bases.

__device__ __forceinline__ void stage_A(unsigned short* Ab,
                                        const float* __restrict__ A, int lda,
                                        int bm, int k0, int M, int tid) {
#pragma unroll
    for (int t = 0; t < 2; ++t) {
        int idx = tid + t * 256;
        int r  = idx >> 3;          // row 0..63
        int c4 = (idx & 7) * 4;     // k   0,4,..,28
        float4 v = make_float4(0.f, 0.f, 0.f, 0.f);
        int gr = bm + r;
        if (gr < M)
            v = *(const float4*)(A + (long)gr * lda + (k0 + c4));
        int sub  = r >> 4;
        int lane = (r & 15) + ((c4 >= 16) ? 16 : 0);
        int e    = c4 & 15;
        *(unsigned long long*)(Ab + ((sub * 32 + lane) * 16 + e)) = pack4_bf16(v);
    }
}

// B tile (no transpose): [k][n] 32 x 64, row-major ldb. float4 along n.
__device__ __forceinline__ void stage_B(unsigned short* Bb,
                                        const float* __restrict__ Bm, int ldb,
                                        int bn, int k0, int N, int tid) {
#pragma unroll
    for (int t = 0; t < 2; ++t) {
        int idx = tid + t * 256;
        int r  = idx >> 4;          // k 0..31
        int c4 = (idx & 15) * 4;    // n 0,4,..,60
        float4 v = make_float4(0.f, 0.f, 0.f, 0.f);
        if (bn + c4 + 4 <= N)
            v = *(const float4*)(Bm + (long)(k0 + r) * ldb + (bn + c4));
        int e  = r & 15;
        int hi = (r >= 16) ? 16 : 0;
        int sub0 = c4 >> 4;
        int l0   = (c4 & 15) + hi;
        unsigned short* dst = Bb + (sub0 * 32 + l0) * 16 + e;
        dst[0 * 16]  = f32_to_bf16(v.x);   // consecutive n -> lane+1 (16 halves)
        dst[1 * 16]  = f32_to_bf16(v.y);
        dst[2 * 16]  = f32_to_bf16(v.z);
        dst[3 * 16]  = f32_to_bf16(v.w);
    }
}

// B tile (transB): logical B[k][n] = Bm[n*ldb + k]. float4 along k ->
// contiguous e in one lane -> packed ds_store_b64.
__device__ __forceinline__ void stage_Bt(unsigned short* Bb,
                                         const float* __restrict__ Bm, int ldb,
                                         int bn, int k0, int N, int tid) {
#pragma unroll
    for (int t = 0; t < 2; ++t) {
        int idx = tid + t * 256;
        int r  = idx >> 3;          // n 0..63
        int c4 = (idx & 7) * 4;     // k 0,4,..,28
        float4 v = make_float4(0.f, 0.f, 0.f, 0.f);
        if (bn + r < N)
            v = *(const float4*)(Bm + (long)(bn + r) * ldb + (k0 + c4));
        int sub  = r >> 4;
        int lane = (r & 15) + ((c4 >= 16) ? 16 : 0);
        int e    = c4 & 15;
        *(unsigned long long*)(Bb + ((sub * 32 + lane) * 16 + e)) = pack4_bf16(v);
    }
}

// ---- generic batched WMMA GEMM --------------------------------------------
// C[z] = alpha * A[z] @ B[z] (+ C_old if accum) (+ bias[col] if bias)
__global__ __launch_bounds__(256) void gemm_wmma(
    const float* __restrict__ A, int lda, long sAo, long sAi,
    const float* __restrict__ Bm, int ldb, long sBo, long sBi, int transB,
    float* __restrict__ C, int ldc, long sCo, long sCi,
    const float* __restrict__ bias, int subBatch,
    int M, int N, int K, float alpha, int accum)
{
    __shared__ __align__(16) unsigned short Asm[2][TILE_HALVES];
    __shared__ __align__(16) unsigned short Bsm[2][TILE_HALVES];

    const int z = blockIdx.z;
    const long bo = z / subBatch, bi = z % subBatch;
    A  += bo * sAo + bi * sAi;
    Bm += bo * sBo + bi * sBi;
    C  += bo * sCo + bi * sCi;

    const int bm = blockIdx.y * 64, bn = blockIdx.x * 64;
    const int tid  = threadIdx.x;
    const int wid  = tid >> 5, lane = tid & 31;
    const int lr   = lane & 15, hi = lane >> 4;
    const int wm   = (wid & 3) * 16;        // 0,16,32,48
    const int wn   = (wid >> 2) * 32;       // 0,32

    const int nk = K >> 5;                  // K % 32 == 0 by construction

    f32x8 acc0 = {}, acc1 = {};

    // prologue: stage tile 0 into buffer 0
    stage_A(Asm[0], A, lda, bm, 0, M, tid);
    if (transB) stage_Bt(Bsm[0], Bm, ldb, bn, 0, N, tid);
    else        stage_B (Bsm[0], Bm, ldb, bn, 0, N, tid);
    __syncthreads();

    for (int kt = 0; kt < nk; ++kt) {
        const int cur = kt & 1;
        if (kt + 1 < nk) {
            const int k0 = (kt + 1) * 32;
            stage_A(Asm[cur ^ 1], A, lda, bm, k0, M, tid);
            if (transB) stage_Bt(Bsm[cur ^ 1], Bm, ldb, bn, k0, N, tid);
            else        stage_B (Bsm[cur ^ 1], Bm, ldb, bn, k0, N, tid);
        }

        const unsigned short* Ab = Asm[cur];
        const unsigned short* Bb = Bsm[cur];
        const int asub = wm >> 4;
        const int bsub = wn >> 4;

        bf16x16 a  = __builtin_bit_cast(bf16x16,
                        *(const s16x16*)(Ab + (asub * 32 + lane) * 16));
        bf16x16 b0 = __builtin_bit_cast(bf16x16,
                        *(const s16x16*)(Bb + ((bsub + 0) * 32 + lane) * 16));
        bf16x16 b1 = __builtin_bit_cast(bf16x16,
                        *(const s16x16*)(Bb + ((bsub + 1) * 32 + lane) * 16));

        acc0 = __builtin_amdgcn_wmma_f32_16x16x32_bf16(
                   false, a, false, b0, (short)0, acc0, false, false);
        acc1 = __builtin_amdgcn_wmma_f32_16x16x32_bf16(
                   false, a, false, b1, (short)0, acc1, false, false);

        __syncthreads();
    }

    // C layout: VGPR r -> row = r + hi*8, col = lane & 15.
#pragma unroll
    for (int r = 0; r < 8; ++r) {
        int row  = bm + wm + r + hi * 8;
        int col0 = bn + wn + lr;
        int col1 = col0 + 16;
        if (row < M) {
            if (col0 < N) {
                float v = alpha * acc0[r];
                if (accum) v += C[(long)row * ldc + col0];
                if (bias)  v += bias[col0];
                C[(long)row * ldc + col0] = v;
            }
            if (col1 < N) {
                float v = alpha * acc1[r];
                if (accum) v += C[(long)row * ldc + col1];
                if (bias)  v += bias[col1];
                C[(long)row * ldc + col1] = v;
            }
        }
    }
}

// ---- elementwise / reduction kernels --------------------------------------

__global__ __launch_bounds__(256) void k_temb(const float* __restrict__ ts,
                                              float* __restrict__ temb) {
    const float TWO_PI = 6.28318530717958647692f;
    for (int i = blockIdx.x * 256 + threadIdx.x; i < BB * HH;
         i += gridDim.x * 256) {
        int b = i / HH, c = i % HH;
        int j = (c < HH / 2) ? c : c - HH / 2;
        float frac = (float)j / (float)(HH / 2 - 1);
        float period = 4e-3f * powf(1000.f, frac);   // MIN_P*(MAX_P/MIN_P)^frac
        float s = TWO_PI / period * ts[b];
        temb[i] = (c < HH / 2) ? sinf(s) : cosf(s);
    }
}

__global__ __launch_bounds__(256) void k_silu(float* __restrict__ x, int n) {
    for (int i = blockIdx.x * 256 + threadIdx.x; i < n; i += gridDim.x * 256) {
        float v = x[i];
        x[i] = v / (1.f + expf(-v));
    }
}

__global__ __launch_bounds__(256) void k_preflen(const unsigned char* __restrict__ mask,
                                                 int* __restrict__ plen) {
    int b = threadIdx.x;
    if (b < BB) {
        int s = 0;
        for (int t = 0; t < PP; ++t) s += (mask[b * PP + t] != 0);
        plen[b] = s;
    }
}

// out[row] = rms(x[row]) * w * (1 + ada[b])   (ada nullable)
__global__ __launch_bounds__(256) void k_rms_ada(const float* __restrict__ x,
                                                 const float* __restrict__ w,
                                                 const float* __restrict__ ada,
                                                 float* __restrict__ out,
                                                 int rowsPerB) {
    __shared__ float red[256];
    const int row = blockIdx.x;
    const float* xr = x + (long)row * HH;
    float ss = 0.f;
    for (int i = threadIdx.x; i < HH; i += 256) { float v = xr[i]; ss += v * v; }
    red[threadIdx.x] = ss;
    __syncthreads();
    for (int o = 128; o > 0; o >>= 1) {
        if (threadIdx.x < o) red[threadIdx.x] += red[threadIdx.x + o];
        __syncthreads();
    }
    float inv = rsqrtf(red[0] / (float)HH + 1e-6f);
    int b = row / rowsPerB;
    for (int i = threadIdx.x; i < HH; i += 256) {
        float g = ada ? (1.f + ada[(long)b * HH + i]) : 1.f;
        out[(long)row * HH + i] = xr[i] * inv * w[i] * g;
    }
}

// RoPE on [T, nh, 256]
__global__ __launch_bounds__(256) void k_rope(float* __restrict__ x, int nh,
                                              const int* __restrict__ plen) {
    int total = TT * nh * (HD / 2);
    for (int i = blockIdx.x * 256 + threadIdx.x; i < total; i += gridDim.x * 256) {
        int j = i % (HD / 2);
        int rest = i / (HD / 2);
        int h = rest % nh;
        int tok = rest / nh;
        int b = tok / SS, s = tok % SS;
        float pos = (float)(plen[b] + s);
        float freq = powf(10000.f, -(2.f * j) / (float)HD);
        float ang = pos * freq;
        float c = cosf(ang), sn = sinf(ang);
        long base = ((long)tok * nh + h) * HD;
        float x1 = x[base + j], x2 = x[base + HD / 2 + j];
        x[base + j]          = x1 * c - x2 * sn;
        x[base + HD / 2 + j] = x2 * c + x1 * sn;
    }
}

// Kc/Vc[b,t,d]: t<P from cache, P<=t<PT from new k/v, PT<=t<SCP zero pad
__global__ __launch_bounds__(256) void k_concat(const float* __restrict__ cache,
                                                int l,
                                                const float* __restrict__ kbuf,
                                                const float* __restrict__ vbuf,
                                                float* __restrict__ Kc,
                                                float* __restrict__ Vc) {
    long total = (long)BB * SCP * HD;
    for (long i = blockIdx.x * 256L + threadIdx.x; i < total;
         i += (long)gridDim.x * 256) {
        int d = (int)(i % HD);
        long rest = i / HD;
        int t = (int)(rest % SCP);
        int b = (int)(rest / SCP);
        float kvv = 0.f, vvv = 0.f;
        if (t < PP) {
            long ko = ((((long)l * 2 + 0) * BB + b) * PP + t) * HD + d;
            long vo = ((((long)l * 2 + 1) * BB + b) * PP + t) * HD + d;
            kvv = cache[ko]; vvv = cache[vo];
        } else if (t < PT) {
            long o = ((long)(b * SS + (t - PP))) * HD + d;
            kvv = kbuf[o]; vvv = vbuf[o];
        }
        Kc[i] = kvv; Vc[i] = vvv;
    }
}

// masked softmax over rows of scores [B*NH*S, SCP]; cols >= PT forced to 0
__global__ __launch_bounds__(256) void k_softmax(float* __restrict__ sc,
                                                 const unsigned char* __restrict__ mask) {
    __shared__ float red[256];
    const int row = blockIdx.x;
    const int b = row / (NH * SS);
    float* r = sc + (long)row * SCP;

    float mx = -3.4e38f;
    for (int t = threadIdx.x; t < SCP; t += 256) {
        float v = r[t];
        bool dead = (t >= PT) || (t < PP && mask[b * PP + t] == 0);
        if (dead) v = -1e9f;
        mx = fmaxf(mx, v);
    }
    red[threadIdx.x] = mx; __syncthreads();
    for (int o = 128; o > 0; o >>= 1) {
        if (threadIdx.x < o) red[threadIdx.x] = fmaxf(red[threadIdx.x], red[threadIdx.x + o]);
        __syncthreads();
    }
    mx = red[0]; __syncthreads();

    float sum = 0.f;
    for (int t = threadIdx.x; t < SCP; t += 256) {
        float v = r[t];
        bool dead = (t >= PT) || (t < PP && mask[b * PP + t] == 0);
        float e = dead ? 0.f : expf(v - mx);
        r[t] = e; sum += e;
    }
    red[threadIdx.x] = sum; __syncthreads();
    for (int o = 128; o > 0; o >>= 1) {
        if (threadIdx.x < o) red[threadIdx.x] += red[threadIdx.x + o];
        __syncthreads();
    }
    float inv = 1.f / red[0];
    for (int t = threadIdx.x; t < SCP; t += 256) r[t] *= inv;
}

__global__ __launch_bounds__(256) void k_gelumul(float* __restrict__ g,
                                                 const float* __restrict__ u,
                                                 long n) {
    for (long i = blockIdx.x * 256L + threadIdx.x; i < n;
         i += (long)gridDim.x * 256) {
        float x = g[i];
        float t = tanhf(0.7978845608028654f * (x + 0.044715f * x * x * x));
        g[i] = 0.5f * x * (1.f + t) * u[i];
    }
}

// ---- host side -------------------------------------------------------------

static inline void gemm(hipStream_t st,
                        const float* A, int lda, long sAo, long sAi,
                        const float* B, int ldb, long sBo, long sBi, int transB,
                        float* C, int ldc, long sCo, long sCi,
                        const float* bias, int subBatch,
                        int M, int N, int K, float alpha, int accum, int batches) {
    dim3 grid((N + 63) / 64, (M + 63) / 64, batches);
    gemm_wmma<<<grid, 256, 0, st>>>(A, lda, sAo, sAi, B, ldb, sBo, sBi, transB,
                                    C, ldc, sCo, sCi, bias, subBatch,
                                    M, N, K, alpha, accum);
}

extern "C" void kernel_launch(void* const* d_in, const int* in_sizes, int n_in,
                              void* d_out, int out_size, void* d_ws, size_t ws_size,
                              hipStream_t stream) {
    (void)in_sizes; (void)n_in; (void)out_size; (void)ws_size;

    const float* x_t    = (const float*)d_in[0];
    const float* tstep  = (const float*)d_in[1];
    const unsigned char* pmask = (const unsigned char*)d_in[2];
    const float* cache  = (const float*)d_in[3];
    const float* Wa_in  = (const float*)d_in[4];
    const float* ba_in  = (const float*)d_in[5];
    const float* Wt_in  = (const float*)d_in[6];
    const float* bt_in  = (const float*)d_in[7];
    const float* Wt_out = (const float*)d_in[8];
    const float* bt_out = (const float*)d_in[9];
    const float* Wq     = (const float*)d_in[10];
    const float* Wk     = (const float*)d_in[11];
    const float* Wv     = (const float*)d_in[12];
    const float* Wo     = (const float*)d_in[13];
    const float* Wg     = (const float*)d_in[14];
    const float* Wu     = (const float*)d_in[15];
    const float* Wd     = (const float*)d_in[16];
    const float* n1_w   = (const float*)d_in[17];
    const float* n2_w   = (const float*)d_in[18];
    const float* Wada1  = (const float*)d_in[19];
    const float* Wada2  = (const float*)d_in[20];
    const float* nf_w   = (const float*)d_in[21];
    const float* Wa_out = (const float*)d_in[22];
    const float* ba_out = (const float*)d_in[23];
    float* out = (float*)d_out;

    // workspace layout (floats)
    float* ws = (float*)d_ws;
    long off = 0;
    float* temb = ws + off; off += (long)BB * HH;
    float* t1   = ws + off; off += (long)BB * HH;
    float* tvec = ws + off; off += (long)BB * HH;
    float* ada  = ws + off; off += (long)BB * HH;
    float* x    = ws + off; off += (long)TT * HH;
    float* h    = ws + off; off += (long)TT * HH;
    float* qb   = ws + off; off += (long)TT * NH * HD;
    float* kb   = ws + off; off += (long)TT * HD;
    float* vb   = ws + off; off += (long)TT * HD;
    float* Kc   = ws + off; off += (long)BB * SCP * HD;
    float* Vc   = ws + off; off += (long)BB * SCP * HD;
    float* sc   = ws + off; off += (long)BB * NH * SS * SCP;
    float* ao   = ws + off; off += (long)TT * NH * HD;
    float* gb   = ws + off; off += (long)TT * MLPD;
    float* ub   = ws + off; off += (long)TT * MLPD;
    int*   plen = (int*)(ws + off);

    const float scale = 0.0625f; // HD^-0.5

    // --- time embedding + conditioning MLP ---
    k_temb<<<32, 256, 0, stream>>>(tstep, temb);
    gemm(stream, temb, HH, 0, 0, Wt_in, HH, 0, 0, 0,
         t1, HH, 0, 0, bt_in, 1, BB, HH, HH, 1.f, 0, 1);
    k_silu<<<32, 256, 0, stream>>>(t1, BB * HH);
    gemm(stream, t1, HH, 0, 0, Wt_out, HH, 0, 0, 0,
         tvec, HH, 0, 0, bt_out, 1, BB, HH, HH, 1.f, 0, 1);
    k_silu<<<32, 256, 0, stream>>>(tvec, BB * HH);

    // --- input embed: x = x_t @ Wa_in + ba_in ---
    gemm(stream, x_t, AD, 0, 0, Wa_in, HH, 0, 0, 0,
         x, HH, 0, 0, ba_in, 1, TT, HH, AD, 1.f, 0, 1);

    k_preflen<<<1, 256, 0, stream>>>(pmask, plen);

    for (int l = 0; l < LL; ++l) {
        const float* Wq_l = Wq + (long)l * HH * (NH * HD);
        const float* Wk_l = Wk + (long)l * HH * HD;
        const float* Wv_l = Wv + (long)l * HH * HD;
        const float* Wo_l = Wo + (long)l * (NH * HD) * HH;
        const float* Wg_l = Wg + (long)l * HH * MLPD;
        const float* Wu_l = Wu + (long)l * HH * MLPD;
        const float* Wd_l = Wd + (long)l * MLPD * HH;
        const float* W1_l = Wada1 + (long)l * HH * HH;
        const float* W2_l = Wada2 + (long)l * HH * HH;

        // adaLN-1 + RMS
        gemm(stream, tvec, HH, 0, 0, W1_l, HH, 0, 0, 0,
             ada, HH, 0, 0, nullptr, 1, BB, HH, HH, 1.f, 0, 1);
        k_rms_ada<<<TT, 256, 0, stream>>>(x, n1_w + (long)l * HH, ada, h, SS);

        // QKV projections
        gemm(stream, h, HH, 0, 0, Wq_l, NH * HD, 0, 0, 0,
             qb, NH * HD, 0, 0, nullptr, 1, TT, NH * HD, HH, 1.f, 0, 1);
        gemm(stream, h, HH, 0, 0, Wk_l, HD, 0, 0, 0,
             kb, HD, 0, 0, nullptr, 1, TT, HD, HH, 1.f, 0, 1);
        gemm(stream, h, HH, 0, 0, Wv_l, HD, 0, 0, 0,
             vb, HD, 0, 0, nullptr, 1, TT, HD, HH, 1.f, 0, 1);

        k_rope<<<512, 256, 0, stream>>>(qb, NH, plen);
        k_rope<<<128, 256, 0, stream>>>(kb, 1, plen);

        k_concat<<<1024, 256, 0, stream>>>(cache, l, kb, vb, Kc, Vc);

        // scores[b,h] = Q[b,:,h,:] @ Kc[b]^T * scale   (batched B*NH)
        gemm(stream,
             qb, NH * HD, (long)SS * NH * HD, (long)HD,
             Kc, HD, (long)SCP * HD, 0, /*transB=*/1,
             sc, SCP, (long)NH * SS * SCP, (long)SS * SCP,
             nullptr, NH, SS, SCP, HD, scale, 0, BB * NH);

        k_softmax<<<BB * NH * SS, 256, 0, stream>>>(sc, pmask);

        // o[b,h] = P @ Vc[b]   (batched B*NH), written head-interleaved
        gemm(stream,
             sc, SCP, (long)NH * SS * SCP, (long)SS * SCP,
             Vc, HD, (long)SCP * HD, 0, 0,
             ao, NH * HD, (long)SS * NH * HD, (long)HD,
             nullptr, NH, SS, HD, SCP, 1.f, 0, BB * NH);

        // x += o @ Wo
        gemm(stream, ao, NH * HD, 0, 0, Wo_l, HH, 0, 0, 0,
             x, HH, 0, 0, nullptr, 1, TT, HH, NH * HD, 1.f, 1, 1);

        // adaLN-2 + RMS
        gemm(stream, tvec, HH, 0, 0, W2_l, HH, 0, 0, 0,
             ada, HH, 0, 0, nullptr, 1, BB, HH, HH, 1.f, 0, 1);
        k_rms_ada<<<TT, 256, 0, stream>>>(x, n2_w + (long)l * HH, ada, h, SS);

        // MLP
        gemm(stream, h, HH, 0, 0, Wg_l, MLPD, 0, 0, 0,
             gb, MLPD, 0, 0, nullptr, 1, TT, MLPD, HH, 1.f, 0, 1);
        gemm(stream, h, HH, 0, 0, Wu_l, MLPD, 0, 0, 0,
             ub, MLPD, 0, 0, nullptr, 1, TT, MLPD, HH, 1.f, 0, 1);
        k_gelumul<<<2048, 256, 0, stream>>>(gb, ub, (long)TT * MLPD);
        gemm(stream, gb, MLPD, 0, 0, Wd_l, HH, 0, 0, 0,
             x, HH, 0, 0, nullptr, 1, TT, HH, MLPD, 1.f, 1, 1);
    }

    // final norm + head
    k_rms_ada<<<TT, 256, 0, stream>>>(x, nf_w, nullptr, h, SS);
    gemm(stream, h, HH, 0, 0, Wa_out, AD, 0, 0, 0,
         out, AD, 0, 0, ba_out, 1, TT, AD, HH, 1.f, 0, 1);
}
